// ThorMoE_8899172237666
// MI455X (gfx1250) — matrix operand order repaired and checked by reading, not compiled, not run on hardware
//
#include <hip/hip_runtime.h>
#include <hip/hip_bf16.h>
#include <math.h>

#define Hdim 768
#define Idim 3072
#define Edim 8
#define Tdim 1024
#define PADMAX 1280   // T + E*16 = 1152 max padded rows, with slack

typedef __attribute__((ext_vector_type(16))) __bf16 v16bf;
typedef __attribute__((ext_vector_type(8)))  float  v8f;
typedef __attribute__((ext_vector_type(4)))  unsigned int v4u;
typedef __attribute__((ext_vector_type(8)))  int v8i;
typedef __attribute__((ext_vector_type(4)))  int v4i;

// ---------------------------------------------------------------------------
// Kernel 1: counting-sort tokens by expert. seg[e]..seg[e+1] is expert e's
// 16-padded row range; perm[] maps padded row -> token id (-1 = pad).
// ---------------------------------------------------------------------------
__global__ __launch_bounds__(256) void build_perm_kernel(
    const int* __restrict__ eidx, int* __restrict__ seg, int* __restrict__ perm) {
  __shared__ int cnt[Edim];
  __shared__ int segs[Edim + 1];
  __shared__ int cur[Edim];
  int tid = threadIdx.x;
  if (tid < Edim) { cnt[tid] = 0; cur[tid] = 0; }
  __syncthreads();
  for (int t = tid; t < Tdim; t += 256) atomicAdd(&cnt[eidx[t]], 1);
  __syncthreads();
  if (tid == 0) {
    int off = 0;
    for (int e = 0; e < Edim; ++e) {
      segs[e] = off;
      off += ((cnt[e] + 15) >> 4) << 4;   // pad each expert to multiple of 16
    }
    segs[Edim] = off;
    for (int e = 0; e <= Edim; ++e) seg[e] = segs[e];
  }
  __syncthreads();
  for (int i = tid; i < PADMAX; i += 256) perm[i] = -1;
  __syncthreads();
  for (int t = tid; t < Tdim; t += 256) {
    int e = eidx[t];
    int p = atomicAdd(&cur[e], 1);
    perm[segs[e] + p] = t;
  }
}

__device__ __forceinline__ float gelu_exact(float v) {
  return 0.5f * v * (1.0f + erff(v * 0.70710678118654752f));
}

__device__ __forceinline__ unsigned short bf16_bits(float f) {
  __bf16 h = (__bf16)f;
  return __builtin_bit_cast(unsigned short, h);
}

// Async global->LDS 16-byte copy (GLOBAL_LOAD_ASYNC_TO_LDS_B128, ASYNCcnt).
__device__ __forceinline__ void async_copy_b128(unsigned ldsoff, const void* gaddr) {
  asm volatile("global_load_async_to_lds_b128 %0, %1, off"
               :: "v"(ldsoff), "v"(gaddr) : "memory");
}
__device__ __forceinline__ void wait_async0() {
  asm volatile("s_wait_asynccnt 0x0" ::: "memory");
}
__device__ __forceinline__ int rfl(int v) {
  return __builtin_amdgcn_readfirstlane(v);
}

// TDM gather-mode load: gather 16 rows of `rowElems` f32 (16-bit row indices)
// from tensor base `gbase` (row stride = rowStride elems) into LDS at ldsa.
// D# bitfields per CDNA5 ISA 8.3-8.7.
__device__ __forceinline__ void tdm_gather16_f32(
    unsigned ldsa, const float* gbase, int rowElems, int rowStride,
    int numRows, const int* idx16 /* 8 packed words, uniform */) {
  unsigned long long ga = (unsigned long long)gbase;
  v4u g0;
  g0.x = 1u | (1u << 31);                       // count=1, gather_mode=1 (16-bit idx)
  g0.y = ldsa;                                  // lds_addr
  g0.z = (unsigned)ga;                          // global_addr[31:0]
  g0.w = (unsigned)(ga >> 32) | (2u << 30);     // global_addr[56:32], type=2
  v8i g1;
  g1[0] = (2 << 16);                            // data_size=2 (4B), wg_mask=0
  g1[1] = (rowElems & 0xFFFF) << 16;            // tensor_dim0[15:0]
  g1[2] = ((unsigned)rowElems >> 16) | (numRows << 16);  // td0[31:16], td1[15:0]
  g1[3] = (rowElems << 16);                     // td1[31:16]=0, tile_dim0
  g1[4] = 16;                                   // tile_dim1 = #indices
  g1[5] = rowStride;                            // tensor_dim0_stride[31:0]
  g1[6] = 0;                                    // stride hi / td1_stride (ignored)
  g1[7] = 0;
  v4i g2, g3;
  g2[0] = idx16[0]; g2[1] = idx16[1]; g2[2] = idx16[2]; g2[3] = idx16[3];
  g3[0] = idx16[4]; g3[1] = idx16[5]; g3[2] = idx16[6]; g3[3] = idx16[7];
#if __clang_major__ >= 23
  v8i gz = {};
  __builtin_amdgcn_tensor_load_to_lds(g0, g1, g2, g3, gz, 0);
#else
  __builtin_amdgcn_tensor_load_to_lds(g0, g1, g2, g3, 0);
#endif
}

// ---------------------------------------------------------------------------
// Kernel 2: inter = gelu(x @ W1[e] + b1[e]) for each expert's token group.
// Block = 16 token rows x 256 cols of I; 8 waves x two 16x16 WMMA tiles.
// A tile (16 permuted token rows x 768 f32) staged by ONE TDM gather-mode
// tensor_load_to_lds; B loaded global->reg with cvt (no inter-wave reuse).
// ---------------------------------------------------------------------------
__global__ __launch_bounds__(256) void gemm1_gelu_kernel(
    const float* __restrict__ x, const float* __restrict__ W1,
    const float* __restrict__ b1, const int* __restrict__ seg,
    const int* __restrict__ perm, unsigned short* __restrict__ inter) {
  __shared__ __align__(16) float Af[16 * Hdim];   // 48 KB, f32 A tile
  __shared__ int rtok[16];
  __shared__ int tsrc[16];

  int e = blockIdx.x >> 6;
  int tile_m = blockIdx.x & 63;
  int s0 = seg[e];
  if (tile_m * 16 >= seg[e + 1] - s0) return;     // uniform early exit
  int rowbase = s0 + tile_m * 16;
  int tid = threadIdx.x;
  if (tid < 16) {
    int t = perm[rowbase + tid];
    rtok[tid] = t;
    tsrc[tid] = (t < 0) ? 0 : t;  // alias pad rows to token 0 (outputs masked)
  }
  __syncthreads();

  int lane = tid & 31, wv = tid >> 5;
  if (wv == 0) {
    // wave 0 issues the TDM gather of all 16 token rows, waits TENSORcnt
    int idxw[8];
#pragma unroll
    for (int w = 0; w < 8; ++w)
      idxw[w] = rfl(tsrc[2 * w] | (tsrc[2 * w + 1] << 16));
    tdm_gather16_f32((unsigned)(size_t)&Af[0], x, Hdim, Hdim, Tdim, idxw);
    __builtin_amdgcn_s_wait_tensorcnt(0);
  }
  __syncthreads();

  int half = lane >> 4, mrow = lane & 15;
  int nl = wv * 16 + mrow;                        // local column 0..127
  int n0 = blockIdx.y * 256;
  const float* Wg = W1 + (size_t)e * Hdim * Idim;

  v8f acc0 = {}, acc1 = {};
  for (int k0 = 0; k0 < Hdim; k0 += 32) {         // barrier-free K loop
    // A fragment: lane m = lane&15; VGPR i holds K pair 16*half+2i (contiguous K, same order as B)
    union { v16bf v; __bf16 h[16]; } af;
#pragma unroll
    for (int i = 0; i < 8; ++i) {
      int kb = k0 + half * 16 + (i << 1);
      float2 p = *(const float2*)&Af[mrow * Hdim + kb];
      af.h[2 * i]     = (__bf16)p.x;
      af.h[2 * i + 1] = (__bf16)p.y;
    }
    // B fragments: lane n fixed, 16 consecutive K; direct global loads + cvt
    const float* bp = Wg + (size_t)(k0 + half * 16) * Idim + (n0 + nl);
    __builtin_prefetch(bp + (size_t)32 * Idim, 0, 1);  // next K-step (spec.)
    union { v16bf v; __bf16 h[16]; } bf0, bf1;
#pragma unroll
    for (int j = 0; j < 16; ++j) {
      bf0.h[j] = (__bf16)bp[(size_t)j * Idim];
      bf1.h[j] = (__bf16)bp[(size_t)j * Idim + 128];
    }
    acc0 = __builtin_amdgcn_wmma_f32_16x16x32_bf16(false, af.v, false, bf0.v,
                                                   (short)0, acc0, false, false);
    acc1 = __builtin_amdgcn_wmma_f32_16x16x32_bf16(false, af.v, false, bf1.v,
                                                   (short)0, acc1, false, false);
  }

  int mb = half * 8;
  {
    int n = n0 + nl;
    float bias = b1[e * Idim + n];
#pragma unroll
    for (int r = 0; r < 8; ++r)
      inter[(size_t)(rowbase + mb + r) * Idim + n] =
          bf16_bits(gelu_exact(acc0[r] + bias));
  }
  {
    int n = n0 + 128 + nl;
    float bias = b1[e * Idim + n];
#pragma unroll
    for (int r = 0; r < 8; ++r)
      inter[(size_t)(rowbase + mb + r) * Idim + n] =
          bf16_bits(gelu_exact(acc1[r] + bias));
  }
}

// ---------------------------------------------------------------------------
// Kernel 3: y = inter @ W2[e] + b2[e] + x  (scatter to token order, f32 out)
// K = I = 3072; A (already bf16) async-staged in 512-wide chunks (ASYNCcnt).
// ---------------------------------------------------------------------------
__global__ __launch_bounds__(256) void gemm2_res_kernel(
    const unsigned short* __restrict__ inter, const float* __restrict__ W2,
    const float* __restrict__ b2, const float* __restrict__ x,
    const int* __restrict__ seg, const int* __restrict__ perm,
    float* __restrict__ y) {
  __shared__ __align__(16) __bf16 Achunk[16 * 512];   // 16 KB
  __shared__ int rtok[16];

  int e = blockIdx.x >> 6;
  int tile_m = blockIdx.x & 63;
  int s0 = seg[e];
  if (tile_m * 16 >= seg[e + 1] - s0) return;
  int rowbase = s0 + tile_m * 16;
  int tid = threadIdx.x;
  if (tid < 16) rtok[tid] = perm[rowbase + tid];
  __syncthreads();

  int lane = tid & 31, wv = tid >> 5;
  int half = lane >> 4, mrow = lane & 15;
  int nl = wv * 16 + mrow;
  int n0 = blockIdx.y * 256;
  const float* Wg = W2 + (size_t)e * Idim * Hdim;
  unsigned ldsbase = (unsigned)(size_t)&Achunk[0];

  v8f acc0 = {}, acc1 = {};
  for (int kc = 0; kc < Idim; kc += 512) {
    __syncthreads();                    // previous chunk's readers done
    // Async-stage A chunk: 16 x 512 bf16 = 1024 16B chunks, 4 per thread
#pragma unroll
    for (int it = 0; it < 4; ++it) {
      int c = tid + it * 256;
      int r = c >> 6;                   // 64 chunks per row (512*2/16)
      int q = c & 63;
      async_copy_b128(ldsbase + (unsigned)(r * 512 + q * 8) * 2,
                      inter + (size_t)(rowbase + r) * Idim + kc + q * 8);
    }
    wait_async0();
    __syncthreads();

    for (int ks = 0; ks < 512; ks += 32) {
      union { unsigned int u[8]; v16bf v; } af;
#pragma unroll
      for (int i = 0; i < 8; ++i) {
        int kb = ks + half * 16 + (i << 1);
        af.u[i] = *(const unsigned int*)&Achunk[mrow * 512 + kb];
      }
      const float* bp = Wg + (size_t)(kc + ks + half * 16) * Hdim + (n0 + nl);
      __builtin_prefetch(bp + (size_t)32 * Hdim, 0, 1);
      union { v16bf v; __bf16 h[16]; } bf0, bf1;
#pragma unroll
      for (int j = 0; j < 16; ++j) {
        bf0.h[j] = (__bf16)bp[(size_t)j * Hdim];
        bf1.h[j] = (__bf16)bp[(size_t)j * Hdim + 128];
      }
      acc0 = __builtin_amdgcn_wmma_f32_16x16x32_bf16(false, af.v, false, bf0.v,
                                                     (short)0, acc0, false, false);
      acc1 = __builtin_amdgcn_wmma_f32_16x16x32_bf16(false, af.v, false, bf1.v,
                                                     (short)0, acc1, false, false);
    }
  }

  int mb = half * 8;
  float bias0 = b2[e * Hdim + n0 + nl];
  float bias1 = b2[e * Hdim + n0 + 128 + nl];
#pragma unroll
  for (int r = 0; r < 8; ++r) {
    int t = rtok[mb + r];
    if (t >= 0) {
      size_t o0 = (size_t)t * Hdim + (n0 + nl);
      y[o0] = acc0[r] + bias0 + x[o0];
      size_t o1 = (size_t)t * Hdim + (n0 + 128 + nl);
      y[o1] = acc1[r] + bias1 + x[o1];
    }
  }
}

// ---------------------------------------------------------------------------
// Kernel 4: in-place LayerNorm over H=768 per token (one block per token)
// ---------------------------------------------------------------------------
__global__ __launch_bounds__(256) void ln_inplace_kernel(
    float* __restrict__ y, const float* __restrict__ gamma,
    const float* __restrict__ beta) {
  int t = blockIdx.x, tid = threadIdx.x;
  float v[3], s = 0.f, s2 = 0.f;
#pragma unroll
  for (int i = 0; i < 3; ++i) {
    v[i] = y[(size_t)t * Hdim + tid + i * 256];
    s += v[i]; s2 += v[i] * v[i];
  }
  __shared__ float red[16];
  for (int o = 16; o > 0; o >>= 1) { s += __shfl_xor(s, o); s2 += __shfl_xor(s2, o); }
  int lane = tid & 31, wv = tid >> 5;
  if (lane == 0) { red[wv] = s; red[8 + wv] = s2; }
  __syncthreads();
  if (wv == 0) {
    float a = (lane < 8) ? red[lane] : 0.f;
    float b = (lane < 8) ? red[8 + lane] : 0.f;
    for (int o = 4; o > 0; o >>= 1) { a += __shfl_xor(a, o); b += __shfl_xor(b, o); }
    if (lane == 0) { red[0] = a; red[1] = b; }
  }
  __syncthreads();
  float mu = red[0] * (1.0f / Hdim);
  float var = red[1] * (1.0f / Hdim) - mu * mu;
  float rs = rsqrtf(var + 1e-12f);
#pragma unroll
  for (int i = 0; i < 3; ++i) {
    int h = tid + i * 256;
    y[(size_t)t * Hdim + h] = (v[i] - mu) * rs * gamma[h] + beta[h];
  }
}

// ---------------------------------------------------------------------------
extern "C" void kernel_launch(void* const* d_in, const int* in_sizes, int n_in,
                              void* d_out, int out_size, void* d_ws, size_t ws_size,
                              hipStream_t stream) {
  const float* x     = (const float*)d_in[0];
  const int*   eidx  = (const int*)d_in[1];
  const float* W1    = (const float*)d_in[2];
  const float* b1    = (const float*)d_in[3];
  const float* W2    = (const float*)d_in[4];
  const float* b2    = (const float*)d_in[5];
  const float* gamma = (const float*)d_in[6];
  const float* beta  = (const float*)d_in[7];
  float* out = (float*)d_out;

  int* seg  = (int*)d_ws;                  // [E+1]
  int* perm = seg + 16;                    // [PADMAX]
  unsigned short* inter = (unsigned short*)((char*)d_ws + 65536);  // bf16 [PADMAX][I]

  build_perm_kernel<<<1, 256, 0, stream>>>(eidx, seg, perm);
  gemm1_gelu_kernel<<<dim3(Edim * 64, Idim / 256), 256, 0, stream>>>(
      x, W1, b1, seg, perm, inter);
  gemm2_res_kernel<<<dim3(Edim * 64, Hdim / 256), 256, 0, stream>>>(
      inter, W2, b2, x, seg, perm, out);
  ln_inplace_kernel<<<Tdim, 256, 0, stream>>>(out, gamma, beta);
}
